// HireModule_1503238553619
// MI455X (gfx1250) — matrix operand order repaired and checked by reading, not compiled, not run on hardware
//
#include <hip/hip_runtime.h>
#include <hip/hip_bf16.h>

typedef _Float16 h16;
typedef __attribute__((ext_vector_type(16))) _Float16 v16h;
typedef __attribute__((ext_vector_type(8)))  _Float16 v8h;
typedef __attribute__((ext_vector_type(8)))  float    v8f;

#define B_   8
#define H_   225
#define W_   225
#define C_   192
#define HW_  (H_*W_)           /* 50625 */
#define RPB_ 25425             /* 113*225 rows per batch in shifted branches */
#define NR_HW_ (B_*RPB_)       /* 203400 */
#define NR_F_  (B_*HW_)        /* 405000 */
#define NTOT_  (B_*H_*W_*C_)   /* 77760000 */

__device__ __forceinline__ v8f wmma16(v16h a, v16h b, v8f c) {
  return __builtin_amdgcn_wmma_f32_16x16x32_f16(false, a, false, b, (short)0, c, false, false);
}

// A-operand: lane (m = lane&15, half = lane>>4) needs two contiguous 8-elem runs:
// K = [Kbase, Kbase+8) and [Kbase+16, Kbase+24) with Kbase = kb*32 + half*8.
__device__ __forceinline__ v16h loadA_g(const float* p0, const float* p1) {
  float4 a0 = *(const float4*)p0;
  float4 a1 = *(const float4*)(p0 + 4);
  float4 b0 = *(const float4*)p1;
  float4 b1 = *(const float4*)(p1 + 4);
  v16h r;
  r[0]=(h16)a0.x; r[1]=(h16)a0.y; r[2]=(h16)a0.z; r[3]=(h16)a0.w;
  r[4]=(h16)a1.x; r[5]=(h16)a1.y; r[6]=(h16)a1.z; r[7]=(h16)a1.w;
  r[8]=(h16)b0.x; r[9]=(h16)b0.y; r[10]=(h16)b0.z; r[11]=(h16)b0.w;
  r[12]=(h16)b1.x; r[13]=(h16)b1.y; r[14]=(h16)b1.z; r[15]=(h16)b1.w;
  return r;
}

__device__ __forceinline__ v16h loadA_lds(const h16* p0, const h16* p1) {
  v8h lo = *(const v8h*)p0;
  v8h hi = *(const v8h*)p1;
  return __builtin_shufflevector(lo, hi, 0,1,2,3,4,5,6,7,8,9,10,11,12,13,14,15);
}

// Row R of the rearranged tensor maps to two x rows; outputs scatter back to the same rows.
__device__ __forceinline__ void decodeHW(int R, int isH, int &b, int &o0, int &o1) {
  b = R / RPB_; int rem = R - b*RPB_;
  if (isH) {
    int r = rem / W_, q = rem - r*W_;
    int x0 = r + 223; if (x0 >= H_) x0 -= H_;
    int x1 = r + 111;                      // r<=112 -> never wraps
    o0 = ((b*H_ + x0)*W_ + q)*C_;
    o1 = ((b*H_ + x1)*W_ + q)*C_;
  } else {
    int hh = rem / 113, rw = rem - hh*113;
    int c0 = rw + 223; if (c0 >= W_) c0 -= W_;
    int c1 = rw + 111;
    int rb = (b*H_ + hh)*W_;
    o0 = (rb + c0)*C_;
    o1 = (rb + c1)*C_;
  }
}

// ---------------------------------------------------------------------------
// Shifted-branch kernel: gather 2 rows -> (16x384)@(384x96) -> LN -> ReLU ->
// (16x96)@(96x384) -> scatter f16 + per-(b,c) sum accumulation.
// ---------------------------------------------------------------------------
__global__ __launch_bounds__(256) void shift_mlp_kernel(
    const float* __restrict__ x,  const float* __restrict__ w1,
    const float* __restrict__ gv, const float* __restrict__ bv,
    const float* __restrict__ w2, const float* __restrict__ b2,
    h16* __restrict__ outH, float* __restrict__ sumOut, int isH)
{
  extern __shared__ __attribute__((aligned(32))) char smem[];
  h16*   w1p   = (h16*)smem;               // 36864 h16  (B-layout packed wh1)
  h16*   w2p   = (h16*)(smem + 73728);     // 36864 h16
  h16*   stage = (h16*)(smem + 147456);    // 8*16*96 h16
  float* gl    = (float*)(smem + 172032);  // 96
  float* bl    = (float*)(smem + 172416);  // 96
  float* b2l   = (float*)(smem + 172800);  // 384
  float* sums  = (float*)(smem + 174336);  // 1536

  const int tid = threadIdx.x;
  // Pack weights into per-lane-contiguous B-operand layout:
  // entry[(kb*NT + t)*32 + lane][e] = W[kb*32 + (lane>>4)*8 + (e&7) + (e>>3)*16][t*16 + (lane&15)]
  for (int i = tid; i < 36864; i += 256) {
    int e = i & 15, ln = (i >> 4) & 31, tk = i >> 9;
    int t = tk % 6, kb = tk / 6;
    int K = kb*32 + ((ln >> 4) << 3) + (e & 7) + ((e >> 3) << 4);
    int n = t*16 + (ln & 15);
    w1p[i] = (h16)w1[K*96 + n];
  }
  for (int i = tid; i < 36864; i += 256) {
    int e = i & 15, ln = (i >> 4) & 31, tk = i >> 9;
    int t = tk % 24, kb = tk / 24;
    int K = kb*32 + ((ln >> 4) << 3) + (e & 7) + ((e >> 3) << 4);
    int n = t*16 + (ln & 15);
    w2p[i] = (h16)w2[K*384 + n];
  }
  for (int i = tid; i < 96;   i += 256) { gl[i] = gv[i]; bl[i] = bv[i]; }
  for (int i = tid; i < 384;  i += 256) b2l[i] = b2[i];
  for (int i = tid; i < 1536; i += 256) sums[i] = 0.f;
  __syncthreads();

  const int lane = tid & 31, wave = tid >> 5;
  const int half = lane >> 4, col = lane & 15;
  const v8f vzero = {0.f,0.f,0.f,0.f,0.f,0.f,0.f,0.f};
  const int numRB = (NR_HW_ + 127) >> 7;

  for (int rb = blockIdx.x; rb < numRB; rb += gridDim.x) {
    const int base = rb*128 + wave*16;
    int Rm = base + col; if (Rm >= NR_HW_) Rm = NR_HW_ - 1;
    int bm, am0, am1; decodeHW(Rm, isH, bm, am0, am1);

    int b8[8], o08[8], o18[8]; unsigned vmask = 0;
#pragma unroll
    for (int i = 0; i < 8; ++i) {
      int Ri = base + i + half*8;
      if (Ri < NR_HW_) vmask |= (1u << i); else Ri = NR_HW_ - 1;
      decodeHW(Ri, isH, b8[i], o08[i], o18[i]);
    }

    // GEMM1: concat(xrow0,xrow1) (16x384) @ wh1 (384x96)
    v8f acc[6];
#pragma unroll
    for (int t = 0; t < 6; ++t) acc[t] = vzero;
#pragma unroll
    for (int kb = 0; kb < 12; ++kb) {
      int K1 = kb*32 + half*8;
      const float* p0 = (K1 < 192) ? (x + am0 + K1) : (x + am1 + (K1 - 192));
      v16h a = loadA_g(p0, p0 + 16);
#pragma unroll
      for (int t = 0; t < 6; ++t) {
        v16h bmat = *(const v16h*)&w1p[((kb*6 + t)*32 + lane)*16];
        acc[t] = wmma16(a, bmat, acc[t]);
      }
    }

    // LayerNorm(96) + ReLU -> stage(f16).  Row m of D lives in VGPR (m&7) of
    // the 16 lanes with half = m>>3; reduce across those lanes via shfl_xor.
    float ps[8], qs[8];
#pragma unroll
    for (int i = 0; i < 8; ++i) { ps[i] = 0.f; qs[i] = 0.f; }
#pragma unroll
    for (int t = 0; t < 6; ++t)
#pragma unroll
      for (int i = 0; i < 8; ++i) { float v = acc[t][i]; ps[i] += v; qs[i] += v*v; }
#pragma unroll
    for (int m = 1; m <= 8; m <<= 1)
#pragma unroll
      for (int i = 0; i < 8; ++i) {
        ps[i] += __shfl_xor(ps[i], m, 32);
        qs[i] += __shfl_xor(qs[i], m, 32);
      }
    float mean[8], rstd[8];
#pragma unroll
    for (int i = 0; i < 8; ++i) {
      mean[i] = ps[i] * (1.f/96.f);
      float var = qs[i] * (1.f/96.f) - mean[i]*mean[i];
      rstd[i] = rsqrtf(var + 1e-6f);
    }
#pragma unroll
    for (int t = 0; t < 6; ++t) {
      int n = t*16 + col;
      float g = gl[n], bb = bl[n];
#pragma unroll
      for (int i = 0; i < 8; ++i) {
        float v = (acc[t][i] - mean[i]) * rstd[i] * g + bb;
        v = fmaxf(v, 0.f);
        stage[(wave*16 + i + half*8)*96 + n] = (h16)v;
      }
    }
    __syncthreads();

    // GEMM2: (16x96) @ wh2 (96x384), N split into 4 groups of 6 tiles.
#pragma unroll 1
    for (int gsel = 0; gsel < 4; ++gsel) {
      v8f a2acc[6];
#pragma unroll
      for (int t = 0; t < 6; ++t) a2acc[t] = vzero;
#pragma unroll
      for (int kb = 0; kb < 3; ++kb) {
        int K1 = kb*32 + half*8;
        const h16* p0 = &stage[(wave*16 + col)*96 + K1];
        v16h a2 = loadA_lds(p0, p0 + 16);
#pragma unroll
        for (int tt = 0; tt < 6; ++tt) {
          int t2 = gsel*6 + tt;
          v16h bmat = *(const v16h*)&w2p[((kb*24 + t2)*32 + lane)*16];
          a2acc[tt] = wmma16(a2, bmat, a2acc[tt]);
        }
      }
#pragma unroll
      for (int tt = 0; tt < 6; ++tt) {
        int n = (gsel*6 + tt)*16 + col;
        float bias = b2l[n];
        int cch = n, useO1 = 0;
        if (n >= 192) { cch = n - 192; useO1 = 1; }
        float sA = 0.f, sB = 0.f;
        int bA = b8[0], bB = b8[7];
#pragma unroll
        for (int i = 0; i < 8; ++i) {
          if (vmask & (1u << i)) {
            float v = a2acc[tt][i] + bias;
            outH[(useO1 ? o18[i] : o08[i]) + cch] = (h16)v;
            if (b8[i] == bA) sA += v; else sB += v;
          }
        }
        atomicAdd(&sums[bA*192 + cch], sA);
        if (sB != 0.f) atomicAdd(&sums[bB*192 + cch], sB);
      }
    }
    __syncthreads();
  }
  __syncthreads();
  for (int i = tid; i < 1536; i += 256) atomicAdd(&sumOut[i], sums[i]);
}

// ---------------------------------------------------------------------------
// Per-(b,c) spatial sum of x (c-branch mean is linear: mean(x@wc)=mean(x)@wc)
// ---------------------------------------------------------------------------
__global__ void xsum_kernel(const float* __restrict__ x, float* __restrict__ sumX) {
  int b = blockIdx.x, slab = blockIdx.y, c = threadIdx.x;   // blockDim.x == 192
  int s0 = slab * 1583;
  int s1 = s0 + 1583; if (s1 > HW_) s1 = HW_;
  float acc = 0.f;
  const float* p = x + ((size_t)b*HW_ + s0)*C_ + c;
  for (int s = s0; s < s1; ++s) { acc += *p; p += C_; }
  atomicAdd(&sumX[b*C_ + c], acc);
}

// ---------------------------------------------------------------------------
// Tiny attention MLP: a = mean(h+w+c) -> gelu(a@wa1+ba1)@wa2+ba2 -> softmax3
// ---------------------------------------------------------------------------
__global__ __launch_bounds__(256) void attn_kernel(
    const float* __restrict__ sumX, const float* __restrict__ sumH, const float* __restrict__ sumW,
    const float* __restrict__ wc, const float* __restrict__ bc,
    const float* __restrict__ wa1, const float* __restrict__ ba1,
    const float* __restrict__ wa2, const float* __restrict__ ba2,
    float* __restrict__ amixOut)
{
  __shared__ float amean[1536];
  __shared__ float t1[384];
  const int tid = threadIdx.x;
  const float inv = 1.f / (float)HW_;
  for (int i = tid; i < 1536; i += 256) {
    int b = i / 192, c = i - b*192;
    float acc = bc[c];
    for (int k = 0; k < 192; ++k) acc += (sumX[b*192 + k] * inv) * wc[k*192 + c];
    amean[i] = acc + (sumH[i] + sumW[i]) * inv;
  }
  __syncthreads();
  for (int i = tid; i < 384; i += 256) {
    int b = i / 48, j = i - b*48;
    float acc = ba1[j];
    for (int c = 0; c < 192; ++c) acc += amean[b*192 + c] * wa1[c*48 + j];
    t1[i] = 0.5f*acc*(1.f + tanhf(0.7978845608028654f*(acc + 0.044715f*acc*acc*acc)));
  }
  __syncthreads();
  for (int i = tid; i < 1536; i += 256) {
    int b = i / 192, c = i - b*192;
    float e0 = ba2[3*c+0], e1 = ba2[3*c+1], e2 = ba2[3*c+2];
    for (int j = 0; j < 48; ++j) {
      float t = t1[b*48 + j];
      e0 += t * wa2[j*576 + 3*c + 0];
      e1 += t * wa2[j*576 + 3*c + 1];
      e2 += t * wa2[j*576 + 3*c + 2];
    }
    float mx = fmaxf(e0, fmaxf(e1, e2));
    float s0 = expf(e0 - mx), s1 = expf(e1 - mx), s2 = expf(e2 - mx);
    float den = 1.f / (s0 + s1 + s2);
    amixOut[0*1536 + i] = s0*den;
    amixOut[1*1536 + i] = s1*den;
    amixOut[2*1536 + i] = s2*den;
  }
}

// ---------------------------------------------------------------------------
// Final kernel: c = x@wc+bc (WMMA), mix = h*a0+w*a1+c*a2, out = mix@wo+bo (WMMA)
// ---------------------------------------------------------------------------
__global__ __launch_bounds__(256) void final_kernel(
    const float* __restrict__ x,  const float* __restrict__ wc, const float* __restrict__ bc,
    const float* __restrict__ wo, const float* __restrict__ bo,
    const h16* __restrict__ hbuf, const h16* __restrict__ wbuf,
    const float* __restrict__ amix, float* __restrict__ out)
{
  extern __shared__ __attribute__((aligned(32))) char smem[];
  h16*   wcp   = (h16*)smem;               // 36864 h16
  h16*   wop   = (h16*)(smem + 73728);     // 36864 h16
  h16*   stage = (h16*)(smem + 147456);    // 8*16*192 h16
  float* bcl   = (float*)(smem + 196608);  // 192
  float* bol   = (float*)(smem + 197376);  // 192
  float* aml   = (float*)(smem + 198144);  // 4608

  const int tid = threadIdx.x;
  for (int i = tid; i < 36864; i += 256) {
    int e = i & 15, ln = (i >> 4) & 31, tk = i >> 9;
    int t = tk % 12, kb = tk / 12;
    int K = kb*32 + ((ln >> 4) << 3) + (e & 7) + ((e >> 3) << 4);
    int n = t*16 + (ln & 15);
    wcp[i] = (h16)wc[K*192 + n];
    wop[i] = (h16)wo[K*192 + n];
  }
  for (int i = tid; i < 192;  i += 256) { bcl[i] = bc[i]; bol[i] = bo[i]; }
  for (int i = tid; i < 4608; i += 256) aml[i] = amix[i];
  __syncthreads();

  const int lane = tid & 31, wave = tid >> 5;
  const int half = lane >> 4, col = lane & 15;
  const v8f vzero = {0.f,0.f,0.f,0.f,0.f,0.f,0.f,0.f};
  const int numRB = (NR_F_ + 127) >> 7;

  for (int rb = blockIdx.x; rb < numRB; rb += gridDim.x) {
    const int base = rb*128 + wave*16;
    int Rm = base + col; if (Rm >= NR_F_) Rm = NR_F_ - 1;
    const float* xrow = x + (size_t)Rm * C_;

    int off8[8], b8[8]; unsigned vmask = 0;
#pragma unroll
    for (int i = 0; i < 8; ++i) {
      int Ri = base + i + half*8;
      if (Ri < NR_F_) vmask |= (1u << i); else Ri = NR_F_ - 1;
      off8[i] = Ri * C_;
      b8[i] = Ri / HW_;
    }

    // GEMM1: c = x @ wc, fused with h/w/c mixing into stage
#pragma unroll 1
    for (int gsel = 0; gsel < 2; ++gsel) {
      v8f acc[6];
#pragma unroll
      for (int t = 0; t < 6; ++t) acc[t] = vzero;
#pragma unroll
      for (int kb = 0; kb < 6; ++kb) {
        int K1 = kb*32 + half*8;
        v16h a = loadA_g(xrow + K1, xrow + K1 + 16);
#pragma unroll
        for (int tt = 0; tt < 6; ++tt) {
          int t = gsel*6 + tt;
          v16h bmat = *(const v16h*)&wcp[((kb*12 + t)*32 + lane)*16];
          acc[tt] = wmma16(a, bmat, acc[tt]);
        }
      }
#pragma unroll
      for (int tt = 0; tt < 6; ++tt) {
        int n = (gsel*6 + tt)*16 + col;
        float bias = bcl[n];
#pragma unroll
        for (int i = 0; i < 8; ++i) {
          float cv = acc[tt][i] + bias;
          float hv = (float)hbuf[off8[i] + n];
          float wv = (float)wbuf[off8[i] + n];
          int bi = b8[i];
          float a0 = aml[0*1536 + bi*192 + n];
          float a1 = aml[1*1536 + bi*192 + n];
          float a2 = aml[2*1536 + bi*192 + n];
          float mv = hv*a0 + wv*a1 + cv*a2;
          stage[(wave*16 + i + half*8)*192 + n] = (h16)mv;
        }
      }
    }
    __syncthreads();

    // GEMM2: out = mix @ wo + bo
#pragma unroll 1
    for (int gsel = 0; gsel < 2; ++gsel) {
      v8f acc[6];
#pragma unroll
      for (int t = 0; t < 6; ++t) acc[t] = vzero;
#pragma unroll
      for (int kb = 0; kb < 6; ++kb) {
        int K1 = kb*32 + half*8;
        const h16* p0 = &stage[(wave*16 + col)*192 + K1];
        v16h a = loadA_lds(p0, p0 + 16);
#pragma unroll
        for (int tt = 0; tt < 6; ++tt) {
          int t = gsel*6 + tt;
          v16h bmat = *(const v16h*)&wop[((kb*12 + t)*32 + lane)*16];
          acc[tt] = wmma16(a, bmat, acc[tt]);
        }
      }
#pragma unroll
      for (int tt = 0; tt < 6; ++tt) {
        int n = (gsel*6 + tt)*16 + col;
        float bias = bol[n];
#pragma unroll
        for (int i = 0; i < 8; ++i)
          if (vmask & (1u << i)) out[off8[i] + n] = acc[tt][i] + bias;
      }
    }
    __syncthreads();
  }
}

extern "C" void kernel_launch(void* const* d_in, const int* in_sizes, int n_in,
                              void* d_out, int out_size, void* d_ws, size_t ws_size,
                              hipStream_t stream) {
  const float* x   = (const float*)d_in[0];
  const float* wh1 = (const float*)d_in[1];
  const float* gh  = (const float*)d_in[2];
  const float* bh  = (const float*)d_in[3];
  const float* wh2 = (const float*)d_in[4];
  const float* bh2 = (const float*)d_in[5];
  const float* ww1 = (const float*)d_in[6];
  const float* gw  = (const float*)d_in[7];
  const float* bw  = (const float*)d_in[8];
  const float* ww2 = (const float*)d_in[9];
  const float* bw2 = (const float*)d_in[10];
  const float* wc  = (const float*)d_in[11];
  const float* bc  = (const float*)d_in[12];
  const float* wa1 = (const float*)d_in[13];
  const float* ba1 = (const float*)d_in[14];
  const float* wa2 = (const float*)d_in[15];
  const float* ba2 = (const float*)d_in[16];
  const float* wo  = (const float*)d_in[17];
  const float* bo  = (const float*)d_in[18];
  float* out = (float*)d_out;

  char* ws = (char*)d_ws;
  h16*   hbuf = (h16*)ws;                            // NTOT_ f16
  h16*   wbuf = (h16*)(ws + (size_t)NTOT_ * 2);      // NTOT_ f16
  float* sumX = (float*)(ws + (size_t)NTOT_ * 4);    // 1536
  float* sumH = sumX + 1536;                         // 1536
  float* sumW = sumH + 1536;                         // 1536
  float* amix = sumW + 1536;                         // 4608

  hipMemsetAsync(sumX, 0, 3 * 1536 * sizeof(float), stream);
  xsum_kernel<<<dim3(8, 32), 192, 0, stream>>>(x, sumX);

  const size_t smemHW = 180480;
  shift_mlp_kernel<<<640, 256, smemHW, stream>>>(x, wh1, gh, bh, wh2, bh2, hbuf, sumH, 1);
  shift_mlp_kernel<<<640, 256, smemHW, stream>>>(x, ww1, gw, bw, ww2, bw2, wbuf, sumW, 0);

  attn_kernel<<<1, 256, 0, stream>>>(sumX, sumH, sumW, wc, bc, wa1, ba1, wa2, ba2, amix);

  const size_t smemF = 216576;
  final_kernel<<<960, 256, smemF, stream>>>(x, wc, bc, wo, bo, hbuf, wbuf, amix, out);
}
